// Me3zaTransformerBlock_73074573574868
// MI455X (gfx1250) — matrix operand order repaired and checked by reading, not compiled, not run on hardware
//
#include <hip/hip_runtime.h>
#include <hip/hip_bf16.h>
#include <math.h>

// ---------------------------------------------------------------------------
// Me3za transformer block on MI455X (gfx1250): bf16 WMMA + async LDS copies.
//   B=1, S=2048, D=1024, H=16, DH=64, F=4096
//   1) weights f32 -> bf16 packed in WMMA-fragment order (one-time, L2-resident)
//   2) h = rms(x,g1) -> bf16
//   3) q/k/v = h @ w{q,k,v}      (double-buffered async WMMA GEMM)
//   4) rope + L2-normalize q,k -> bf16 ; v -> transposed bf16 (Vt = D x S)
//   5) flash attention w/ yat scores; K/V tiles async-staged in LDS,
//      shared by all 8 waves of the block (WMMA QK^T and P*V)
//   6) x2 = x + attn @ wo
//   7) h2 = rms(x2,g2) -> bf16 (+ row ||h2||^2) ; colsq(w_yat)
//   8) dot = h2 @ w_yat ; 9) y = dot^2/(dist2+eps)*alpha -> bf16
//  10) out = x2 + y @ w_out
// ---------------------------------------------------------------------------

#define DEV __device__ __forceinline__

typedef __bf16 bf16_t;
typedef __attribute__((ext_vector_type(16))) __bf16 bf16x16;
typedef __attribute__((ext_vector_type(8)))  float  f32x8;
typedef unsigned int u32;

constexpr int S  = 2048;
constexpr int D  = 1024;
constexpr int H  = 16;
constexpr int F  = 4096;
constexpr int DH = 64;
constexpr float EPS     = 1e-5f;
constexpr float RMS_EPS = 1e-6f;

// ---------------------------------------------------------------- WMMA core
DEV f32x8 wmma_bf16(bf16x16 a, bf16x16 b, f32x8 c) {
  return __builtin_amdgcn_wmma_f32_16x16x32_bf16(
      false, a, false, b, (short)0, c, false, false);
}

// A-matrix fragment (16x32, 16-bit): lane&15 -> row M; lane half selects
// K in {0..15} (lanes 0-15) vs {16..31} (lanes 16-31).
DEV bf16x16 frag_a(const bf16_t* base, int stride, int lane) {
  const bf16_t* rp = base + (size_t)(lane & 15) * stride;
  int khalf = (lane >> 4) * 16;
  bf16x16 v;
#pragma unroll
  for (int i = 0; i < 8; i++) {
    int k = khalf + (i << 1);
    v[2 * i]     = rp[k];
    v[2 * i + 1] = rp[k + 1];
  }
  return v;
}

// B-matrix fragment when this lane's 16 values are contiguous: 2x b128 loads.
DEV bf16x16 frag_lin(const bf16_t* rp) {
  bf16x16 v;
#pragma unroll
  for (int e = 0; e < 16; e++) v[e] = rp[e];
  return v;
}

// ------------------------------------------------- async global->LDS (CDNA5)
// GLOBAL_LOAD_ASYNC_TO_LDS_B128: per-lane global src, per-lane LDS dst,
// tracked by ASYNCcnt (CDNA5 ISA). Generic LDS pointer
// low 32 bits == LDS byte address (ISA §10.2 aperture rules).
DEV void async_cp16(void* lds, const void* g) {
  asm volatile("global_load_async_to_lds_b128 %0, %1, off"
               :: "v"((u32)(size_t)lds), "v"(g) : "memory");
}
DEV void wait_async_4() { asm volatile("s_wait_asynccnt 4" ::: "memory"); }
DEV void wait_async_2() { asm volatile("s_wait_asynccnt 2" ::: "memory"); }
DEV void wait_async_0() { asm volatile("s_wait_asynccnt 0" ::: "memory"); }

// ------------------------------------------------------------ small kernels
DEV float block_reduce_sum_256(float v) {
  __shared__ float red[8];
  int lane = threadIdx.x & 31, wid = threadIdx.x >> 5;
#pragma unroll
  for (int m = 16; m >= 1; m >>= 1) v += __shfl_xor(v, m, 32);
  __syncthreads();
  if (lane == 0) red[wid] = v;
  __syncthreads();
  float r = 0.f;
#pragma unroll
  for (int i = 0; i < 8; i++) r += red[i];
  return r;
}

// one 256-thread block per row
__global__ __launch_bounds__(256)
void k_rmsnorm(const float* __restrict__ x, const float* __restrict__ g,
               bf16_t* __restrict__ out, float* __restrict__ sumsq, int ncols) {
  int row = blockIdx.x;
  const float* xr = x + (size_t)row * ncols;
  float ss = 0.f;
  for (int c = threadIdx.x; c < ncols; c += blockDim.x) {
    float v = xr[c];
    ss = fmaf(v, v, ss);
  }
  ss = block_reduce_sum_256(ss);
  __shared__ float s_r;
  if (threadIdx.x == 0) s_r = rsqrtf(ss / (float)ncols + RMS_EPS);
  __syncthreads();
  float r = s_r;
  bf16_t* orow = out + (size_t)row * ncols;
  float hs = 0.f;
  for (int c = threadIdx.x; c < ncols; c += blockDim.x) {
    float h = xr[c] * r * g[c];
    orow[c] = (bf16_t)h;
    hs = fmaf(h, h, hs);
  }
  if (sumsq) {
    hs = block_reduce_sum_256(hs);
    if (threadIdx.x == 0) sumsq[row] = hs;
  }
}

// Pack f32 weight W[K][N] -> bf16 fragment-ordered tiles:
// tile (tk,tn) of 32x16 -> 512 contiguous bf16, laid out [lane 0..31][16],
// lane = khalf*16 + n_local, element = k&15.
__global__ void k_pack_w(const float* __restrict__ w, bf16_t* __restrict__ out,
                         int log2N) {
  int i = blockIdx.x * blockDim.x + threadIdx.x;  // over K*N
  int N = 1 << log2N;
  int k = i >> log2N, n = i & (N - 1);
  int tk = k >> 5, kl = k & 31;
  int tn = n >> 4, nl = n & 15;
  int khalf = kl >> 4, kj = kl & 15;
  size_t dst = ((size_t)tk * (N >> 4) + tn) * 512 + (size_t)(khalf * 16 + nl) * 16 + kj;
  out[dst] = (bf16_t)w[i];
}

// one wave per (row, head): lane j handles pair (j, j+32) of DH=64
__global__ __launch_bounds__(256)
void k_rope_norm(const float* __restrict__ in, bf16_t* __restrict__ out) {
  int wid  = (blockIdx.x * blockDim.x + threadIdx.x) >> 5;
  int lane = threadIdx.x & 31;
  int row  = wid >> 4;
  int head = wid & 15;
  const float* t = in + (size_t)row * D + head * DH;
  float t1 = t[lane], t2 = t[lane + 32];
  float freq = __powf(10000.f, -(float)lane / 32.f);
  float sn, cs;
  __sincosf((float)row * freq, &sn, &cs);
  float o1 = t1 * cs - t2 * sn;
  float o2 = t2 * cs + t1 * sn;
  float nrm = o1 * o1 + o2 * o2;
#pragma unroll
  for (int m = 16; m >= 1; m >>= 1) nrm += __shfl_xor(nrm, m, 32);
  float inv = 1.f / (sqrtf(nrm) + 1e-12f);
  bf16_t* o = out + (size_t)row * D + head * DH;
  o[lane]      = (bf16_t)(o1 * inv);
  o[lane + 32] = (bf16_t)(o2 * inv);
}

// f32 in[R][C] -> bf16 out[C][R] (tiled, both sides coalesced)
__global__ __launch_bounds__(256)
void k_transpose_to_bf16(const float* __restrict__ in, bf16_t* __restrict__ outT,
                         int R, int C) {
  __shared__ float tile[32][33];
  int r0 = blockIdx.y * 32, c0 = blockIdx.x * 32;
  int tx = threadIdx.x & 31, ty = threadIdx.x >> 5;  // 32 x 8
#pragma unroll
  for (int i = ty; i < 32; i += 8)
    tile[i][tx] = in[(size_t)(r0 + i) * C + c0 + tx];
  __syncthreads();
#pragma unroll
  for (int i = ty; i < 32; i += 8)
    outT[(size_t)(c0 + i) * R + r0 + tx] = (bf16_t)tile[tx][i];
}

__global__ void k_colsq(const float* __restrict__ w, float* __restrict__ out) {
  int f = blockIdx.x * blockDim.x + threadIdx.x;
  if (f >= F) return;
  float s = 0.f;
  for (int d = 0; d < D; d++) {
    float v = w[(size_t)d * F + f];
    s = fmaf(v, v, s);
  }
  out[f] = s;
}

__global__ void k_yat(const float* __restrict__ dot, const float* __restrict__ hsq,
                      const float* __restrict__ colsq, bf16_t* __restrict__ y) {
  int i = blockIdx.x * blockDim.x + threadIdx.x;
  int r = i >> 12;
  int c = i & (F - 1);
  float dv = dot[i];
  float dist2 = hsq[r] + colsq[c] - 2.f * dv;
  const float alpha = 64.0f / logf(4097.0f);  // sqrt(F)/log(1+F)
  y[i] = (bf16_t)(dv * dv / (dist2 + EPS) * alpha);
}

// -------------------------------------------------------------- WMMA GEMM
// C[M,N](f32, +opt residual) = A[M,K](bf16 row-major) * Bp(K,N packed bf16)
// 256 thr = 8 waves; block tile 128x128; wave tile 32x64; K-step 32;
// double-buffered GLOBAL_LOAD_ASYNC_TO_LDS_B128 pipeline.
constexpr int AST = 40;  // A LDS row stride (pad; 80B rows, b128-aligned)

__global__ __launch_bounds__(256)
void k_gemm_bf16(const bf16_t* __restrict__ A, const bf16_t* __restrict__ Bp,
                 const float* __restrict__ resid, float* __restrict__ C,
                 int M, int N, int K) {
  __shared__ bf16_t As[2][128][AST];
  __shared__ bf16_t Bs[2][4096];
  int tid = threadIdx.x, lane = tid & 31, wave = tid >> 5;
  int wrow = wave >> 1, wcol = wave & 1;
  int m0 = blockIdx.y * 128, n0 = blockIdx.x * 128;

  f32x8 zero = {};
  f32x8 acc[2][4];
#pragma unroll
  for (int mi = 0; mi < 2; mi++)
#pragma unroll
    for (int ni = 0; ni < 4; ni++) acc[mi][ni] = zero;

  int ar = tid >> 1, ac = (tid & 1) * 16;  // A: 2 thr/row, 16 bf16 each
  const bf16_t* Ag = A + (size_t)(m0 + ar) * K + ac;
  const bf16_t* Bg = Bp + (size_t)(n0 >> 4) * 512 + tid * 8;
  size_t bstep = (size_t)(N >> 4) * 512;  // packed elems per K-tile

  // prefetch tile 0
  async_cp16(&As[0][ar][ac], Ag);
  async_cp16(&As[0][ar][ac + 8], Ag + 8);
  async_cp16(&Bs[0][tid * 8], Bg);
  async_cp16(&Bs[0][tid * 8 + 2048], Bg + 2048);

  int T = K >> 5;
  for (int t = 0; t < T; t++) {
    int cur = t & 1;
    if (t + 1 < T) {
      const bf16_t* Ag2 = Ag + (size_t)(t + 1) * 32;
      const bf16_t* Bg2 = Bg + (size_t)(t + 1) * bstep;
      async_cp16(&As[cur ^ 1][ar][ac], Ag2);
      async_cp16(&As[cur ^ 1][ar][ac + 8], Ag2 + 8);
      async_cp16(&Bs[cur ^ 1][tid * 8], Bg2);
      async_cp16(&Bs[cur ^ 1][tid * 8 + 2048], Bg2 + 2048);
      wait_async_4();   // tile t fully in LDS (in-order returns)
    } else {
      wait_async_0();
    }
    __syncthreads();
    bf16x16 af0 = frag_a(&As[cur][wrow * 32][0], AST, lane);
    bf16x16 af1 = frag_a(&As[cur][wrow * 32 + 16][0], AST, lane);
#pragma unroll
    for (int ni = 0; ni < 4; ni++) {
      bf16x16 bf = frag_lin(&Bs[cur][(wcol * 4 + ni) * 512 + lane * 16]);
      acc[0][ni] = wmma_bf16(af0, bf, acc[0][ni]);
      acc[1][ni] = wmma_bf16(af1, bf, acc[1][ni]);
    }
    __syncthreads();  // all waves done with buf[cur] before t+1 overwrites
  }

  int half = lane >> 4, nn = lane & 15;
  int rbase = m0 + wrow * 32 + half * 8;
  int cbase = n0 + wcol * 64 + nn;
  float* cdst = C + (size_t)rbase * N + cbase;
  if (resid) {
    const float* rsrc = resid + (size_t)rbase * N + cbase;
#pragma unroll
    for (int mi = 0; mi < 2; mi++)
#pragma unroll
      for (int i = 0; i < 8; i++) {
        size_t ro = (size_t)(mi * 16 + i) * N;
#pragma unroll
        for (int ni = 0; ni < 4; ni++)
          cdst[ro + ni * 16] = acc[mi][ni][i] + rsrc[ro + ni * 16];
      }
  } else {
#pragma unroll
    for (int mi = 0; mi < 2; mi++)
#pragma unroll
      for (int i = 0; i < 8; i++) {
        size_t ro = (size_t)(mi * 16 + i) * N;
#pragma unroll
        for (int ni = 0; ni < 4; ni++)
          cdst[ro + ni * 16] = acc[mi][ni][i];
      }
  }
}

// ----------------------------------------------------- flash yat-attention
// grid (S/128, H); block 256 = 8 waves; wave owns 16 q-rows.
// q,k: [S][D] bf16 (rope+normalized). vT: [D][S] bf16 (transposed).
// K/V tiles are async-staged into LDS once per block and shared by all
// waves; waves past their causal window process fully-masked tiles (p=0).
constexpr int KST = 72;  // Ks row stride (144B, b128-aligned)
constexpr int VST = 40;  // Vs row stride (80B, b128-aligned)

__global__ __launch_bounds__(256)
void k_attn(const bf16_t* __restrict__ q, const bf16_t* __restrict__ k,
            const bf16_t* __restrict__ vT, bf16_t* __restrict__ o) {
  __shared__ bf16_t Ks[2][32 * KST];  // [key_local][dh 0..63]
  __shared__ bf16_t Vs[2][64 * VST];  // [dh_local][key 0..31]
  __shared__ bf16_t Ps[8][16 * 34];   // per-wave P (16x32) C->A layout bounce
  int tid = threadIdx.x, lane = tid & 31, wave = tid >> 5;
  int head = blockIdx.y;
  int blockQ = blockIdx.x * 128;
  int q0 = blockQ + wave * 16;
  const bf16_t* qb  = q + (size_t)q0 * D + head * DH;
  const bf16_t* kb  = k + head * DH;
  const bf16_t* vTh = vT + (size_t)head * DH * S;

  bf16x16 qf0 = frag_a(qb, D, lane);
  bf16x16 qf1 = frag_a(qb + 32, D, lane);

  f32x8 zero = {};
  f32x8 acc[4];
  float m[8], l[8];
#pragma unroll
  for (int j = 0; j < 4; j++) acc[j] = zero;
#pragma unroll
  for (int i = 0; i < 8; i++) { m[i] = -INFINITY; l[i] = 0.f; }

  int half = lane >> 4, nn = lane & 15;
  bf16_t* Pw = &Ps[wave][0];

  // per-thread async-copy coordinates (one 16B chunk of K and of V each)
  int krow = tid >> 3, kchunk = (tid & 7) * 8;  // K tile: 32 rows x 64 dh
  int vrow = tid >> 2, vchunk = (tid & 3) * 8;  // V tile: 64 dh  x 32 keys
  const bf16_t* kgsrc = kb + (size_t)krow * D + kchunk;
  const bf16_t* vgsrc = vTh + (size_t)vrow * S + vchunk;
  u32 kdst = (u32)(krow * KST + kchunk);
  u32 vdst = (u32)(vrow * VST + vchunk);

  int nk = (blockQ + 159) >> 5;  // block-uniform tile count (keys<=blockQ+127)

  // prefetch tile 0
  async_cp16(&Ks[0][kdst], kgsrc);
  async_cp16(&Vs[0][vdst], vgsrc);

  for (int t32 = 0; t32 < nk; t32++) {
    int k0 = t32 * 32;
    int cur = t32 & 1;
    if (t32 + 1 < nk) {
      int k0n = k0 + 32;
      async_cp16(&Ks[cur ^ 1][kdst], kgsrc + (size_t)k0n * D);
      async_cp16(&Vs[cur ^ 1][vdst], vgsrc + k0n);
      wait_async_2();
    } else {
      wait_async_0();
    }
    __syncthreads();

    // scores: two 16x16 tiles over DH=64 (fragments from LDS)
    f32x8 st0 = zero, st1 = zero;
    {
      const bf16_t* kr0 = &Ks[cur][(size_t)nn * KST + half * 16];
      const bf16_t* kr1 = &Ks[cur][(size_t)(16 + nn) * KST + half * 16];
      st0 = wmma_bf16(qf0, frag_lin(kr0), st0);
      st0 = wmma_bf16(qf1, frag_lin(kr0 + 32), st0);
      st1 = wmma_bf16(qf0, frag_lin(kr1), st1);
      st1 = wmma_bf16(qf1, frag_lin(kr1 + 32), st1);
    }
    // yat transform + causal mask + online softmax
    float y0[8], y1[8], nm[8];
#pragma unroll
    for (int i = 0; i < 8; i++) {
      int row = q0 + i + half * 8;
      float s0 = st0[i], s1 = st1[i];
      float a0 = s0 * s0 / (2.f * (1.f - s0) + EPS);
      float a1 = s1 * s1 / (2.f * (1.f - s1) + EPS);
      if (k0 + nn > row)      a0 = -INFINITY;
      if (k0 + 16 + nn > row) a1 = -INFINITY;
      y0[i] = a0; y1[i] = a1;
      float mx = fmaxf(a0, a1);
#pragma unroll
      for (int sh = 8; sh >= 1; sh >>= 1) mx = fmaxf(mx, __shfl_xor(mx, sh, 32));
      nm[i] = fmaxf(m[i], mx);
    }
#pragma unroll
    for (int i = 0; i < 8; i++) {
      float p0 = __expf(y0[i] - nm[i]);
      float p1 = __expf(y1[i] - nm[i]);
      float rs = p0 + p1;
#pragma unroll
      for (int sh = 8; sh >= 1; sh >>= 1) rs += __shfl_xor(rs, sh, 32);
      float sc = __expf(m[i] - nm[i]);
      l[i] = l[i] * sc + rs;
      m[i] = nm[i];
#pragma unroll
      for (int j = 0; j < 4; j++) acc[j][i] *= sc;
      int prow = i + half * 8;
      Pw[prow * 34 + nn]      = (bf16_t)p0;
      Pw[prow * 34 + 16 + nn] = (bf16_t)p1;
    }
    // O += P(16x32) * V(32x64); V fragments from LDS
    bf16x16 pf = frag_a(Pw, 34, lane);
#pragma unroll
    for (int j = 0; j < 4; j++) {
      const bf16_t* vr = &Vs[cur][(size_t)(16 * j + nn) * VST + half * 16];
      acc[j] = wmma_bf16(pf, frag_lin(vr), acc[j]);
    }
    __syncthreads();  // all waves done with buf[cur] before t+1 overwrites
  }

  bf16_t* ob = o + (size_t)q0 * D + head * DH;
#pragma unroll
  for (int j = 0; j < 4; j++)
#pragma unroll
    for (int i = 0; i < 8; i++) {
      int row = i + half * 8;
      ob[(size_t)row * D + 16 * j + nn] = (bf16_t)(acc[j][i] / l[i]);
    }
}

// ------------------------------------------------------------------ driver
extern "C" void kernel_launch(void* const* d_in, const int* in_sizes, int n_in,
                              void* d_out, int out_size, void* d_ws, size_t ws_size,
                              hipStream_t stream) {
  const float* x    = (const float*)d_in[0];
  const float* wq   = (const float*)d_in[1];
  const float* wk   = (const float*)d_in[2];
  const float* wv   = (const float*)d_in[3];
  const float* wo   = (const float*)d_in[4];
  const float* g1   = (const float*)d_in[5];
  const float* g2   = (const float*)d_in[6];
  const float* wyat = (const float*)d_in[7];
  const float* wout = (const float*)d_in[8];
  float* out = (float*)d_out;

  char* base = (char*)d_ws;
  size_t off = 0;
  auto alloc = [&](size_t bytes) {
    size_t o = off;
    off += (bytes + 255) & ~(size_t)255;
    return o;
  };

  bf16_t* wq_bf   = (bf16_t*)(base + alloc((size_t)2 * D * D));
  bf16_t* wk_bf   = (bf16_t*)(base + alloc((size_t)2 * D * D));
  bf16_t* wv_bf   = (bf16_t*)(base + alloc((size_t)2 * D * D));
  bf16_t* wo_bf   = (bf16_t*)(base + alloc((size_t)2 * D * D));
  bf16_t* wyat_bf = (bf16_t*)(base + alloc((size_t)2 * D * F));
  bf16_t* wout_bf = (bf16_t*)(base + alloc((size_t)2 * F * D));
  bf16_t* h_bf    = (bf16_t*)(base + alloc((size_t)2 * S * D));  // reused as h2
  // 32MB region: phase A = {qf,kf,vf,q_bf,k_bf}; phase B = dot (S*F f32)
  size_t big = alloc((size_t)4 * S * F);
  float*  qf    = (float*)(base + big);
  float*  kf    = qf + (size_t)S * D;
  float*  vf    = kf + (size_t)S * D;
  bf16_t* q_bf  = (bf16_t*)(vf + (size_t)S * D);
  bf16_t* k_bf  = q_bf + (size_t)S * D;
  float*  dot_f = (float*)(base + big);
  bf16_t* vT_bf   = (bf16_t*)(base + alloc((size_t)2 * S * D));
  bf16_t* attn_bf = (bf16_t*)(base + alloc((size_t)2 * S * D));
  float*  x2      = (float*)(base + alloc((size_t)4 * S * D));
  bf16_t* y_bf    = (bf16_t*)(base + alloc((size_t)2 * S * F));
  float*  hsq     = (float*)(base + alloc((size_t)4 * S));
  float*  colsq   = (float*)(base + alloc((size_t)4 * F));

  // 1) weights -> packed bf16 (one-time)
  k_pack_w<<<(D * D) / 256, 256, 0, stream>>>(wq, wq_bf, 10);
  k_pack_w<<<(D * D) / 256, 256, 0, stream>>>(wk, wk_bf, 10);
  k_pack_w<<<(D * D) / 256, 256, 0, stream>>>(wv, wv_bf, 10);
  k_pack_w<<<(D * D) / 256, 256, 0, stream>>>(wo, wo_bf, 10);
  k_pack_w<<<(D * F) / 256, 256, 0, stream>>>(wyat, wyat_bf, 12);
  k_pack_w<<<(F * D) / 256, 256, 0, stream>>>(wout, wout_bf, 10);

  // 2) h = rms(x, g1)
  k_rmsnorm<<<S, 256, 0, stream>>>(x, g1, h_bf, nullptr, D);

  // 3) q/k/v projections
  dim3 g1024(1024 / 128, S / 128);
  k_gemm_bf16<<<g1024, 256, 0, stream>>>(h_bf, wq_bf, nullptr, qf, S, D, D);
  k_gemm_bf16<<<g1024, 256, 0, stream>>>(h_bf, wk_bf, nullptr, kf, S, D, D);
  k_gemm_bf16<<<g1024, 256, 0, stream>>>(h_bf, wv_bf, nullptr, vf, S, D, D);

  // 4) rope + L2 norm q,k ; transpose v -> Vt (D x S bf16)
  k_rope_norm<<<(S * H) / 8, 256, 0, stream>>>(qf, q_bf);
  k_rope_norm<<<(S * H) / 8, 256, 0, stream>>>(kf, k_bf);
  k_transpose_to_bf16<<<dim3(D / 32, S / 32), 256, 0, stream>>>(vf, vT_bf, S, D);

  // 5) flash yat-attention
  k_attn<<<dim3(S / 128, H), 256, 0, stream>>>(q_bf, k_bf, vT_bf, attn_bf);

  // 6) x2 = x + attn @ wo
  k_gemm_bf16<<<g1024, 256, 0, stream>>>(attn_bf, wo_bf, x, x2, S, D, D);

  // 7) h2 = rms(x2, g2) (+ ||h2||^2) ; column sq-norms of w_yat
  k_rmsnorm<<<S, 256, 0, stream>>>(x2, g2, h_bf, hsq, D);
  k_colsq<<<F / 256, 256, 0, stream>>>(wyat, colsq);

  // 8) dot = h2 @ w_yat
  k_gemm_bf16<<<dim3(F / 128, S / 128), 256, 0, stream>>>(
      h_bf, wyat_bf, nullptr, dot_f, S, F, D);

  // 9) y = dot^2/(dist2+eps)*alpha
  k_yat<<<(S * F) / 256, 256, 0, stream>>>(dot_f, hsq, colsq, y_bf);

  // 10) out = x2 + y @ w_out
  k_gemm_bf16<<<g1024, 256, 0, stream>>>(y_bf, wout_bf, x2, out, S, D, F);

  (void)in_sizes; (void)n_in; (void)out_size; (void)ws_size;
}
